// Transformer_86380382257330
// MI455X (gfx1250) — matrix operand order repaired and checked by reading, compiled, not run on hardware
//
#include <hip/hip_runtime.h>
#include <hip/hip_bf16.h>
#include <math.h>

// ---- problem constants (match reference) ----
#define BB   2
#define TT   1024
#define DD   512
#define HH   8
#define HDD  64
#define FFD  2048
#define LL   6
#define VIN  8192
#define VOUT 8190
#define VOUTP 8192      // VOUT padded to 32
#define MM   (BB * TT)  // 2048 activation rows

typedef __attribute__((ext_vector_type(16))) _Float16 v16h;
typedef __attribute__((ext_vector_type(8)))  _Float16 v8h;
typedef __attribute__((ext_vector_type(8)))  float    v8f;

// ------------------------------------------------------------------
// Direct-from-memory WMMA fragment loads (CDNA5 ISA 7.12.2, wave32).
// A 16x32 f16: lane l -> row m0+(l&15); halves [k0+hs, +7] and [k0+16+hs, +7]
//              (hs = 8 for lanes 16..31)  => two b128 loads.
// B 32x16 f16 from K-contiguous [N,K] storage: lane l -> row n0+(l&15),
//              halves [k0+hs, +7] and [k0+16+hs, +7] (same as A) => two b128 loads.
// ------------------------------------------------------------------
__device__ __forceinline__ v16h frag_a_g(const _Float16* __restrict__ base,
                                         int ld, int row0, int k0) {
  const int l  = threadIdx.x & 31;
  const int hs = (l & 16) ? 8 : 0;
  const _Float16* p = base + (size_t)(row0 + (l & 15)) * ld + k0 + hs;
  v8h lo = *(const v8h*)(p);
  v8h hi = *(const v8h*)(p + 16);
  return __builtin_shufflevector(lo, hi, 0, 1, 2, 3, 4, 5, 6, 7,
                                 8, 9, 10, 11, 12, 13, 14, 15);
}

__device__ __forceinline__ v16h frag_b_g(const _Float16* __restrict__ base,
                                         int ld, int n0, int k0) {
  const int l  = threadIdx.x & 31;
  const int ks = (l & 16) ? 8 : 0;
  const _Float16* p = base + (size_t)(n0 + (l & 15)) * ld + k0 + ks;
  v8h lo = *(const v8h*)(p);
  v8h hi = *(const v8h*)(p + 16);
  return __builtin_shufflevector(lo, hi, 0, 1, 2, 3, 4, 5, 6, 7,
                                 8, 9, 10, 11, 12, 13, 14, 15);
}

#define WMMA(a, b, c) \
  __builtin_amdgcn_wmma_f32_16x16x32_f16(false, (a), false, (b), (short)0, (c), false, false)

// ------------------------------------------------------------------
// Weight prep: WT[Npad, K] f16  <-  W[K, N] f32   (tiled transpose)
// grid = (Npad/32, K/32), block = 256
// ------------------------------------------------------------------
__launch_bounds__(256)
__global__ void transpose_f16(const float* __restrict__ W, _Float16* __restrict__ WT,
                              int K, int N) {
  __shared__ _Float16 t[32][33];
  const int k0 = blockIdx.y * 32, n0 = blockIdx.x * 32;
  const int tx = threadIdx.x & 31, ty = threadIdx.x >> 5;  // 8 rows / pass
#pragma unroll
  for (int i = 0; i < 32; i += 8) {
    const int n = n0 + tx;
    t[ty + i][tx] = (n < N) ? (_Float16)W[(size_t)(k0 + ty + i) * N + n]
                            : (_Float16)0.f;
  }
  __syncthreads();
#pragma unroll
  for (int i = 0; i < 32; i += 8)
    WT[(size_t)(n0 + ty + i) * K + k0 + tx] = t[tx][ty + i];
}

// ------------------------------------------------------------------
// GEMM: C[M,N] = act(A_f16[M,K] x BT_f16[Npad,K]^T + bias)
// block = 128 (4 waves); each wave owns a 64x32 macro-tile (4x2 WMMA tiles)
// grid = (Npad/32, M/256).  vt=1: scatter C as [B,H,HD,T] f16 (V transpose)
// ------------------------------------------------------------------
__launch_bounds__(128)
__global__ void gemm_wmma(const _Float16* __restrict__ A,
                          const _Float16* __restrict__ BT,
                          const float* __restrict__ bias,
                          float* __restrict__ C32,
                          _Float16* __restrict__ C16,
                          int N, int K, int relu, int vt) {
  const int w  = threadIdx.x >> 5;
  const int l  = threadIdx.x & 31;
  const int m0 = blockIdx.y * 256 + w * 64;
  const int n0 = blockIdx.x * 32;
  v8f c[4][2] = {};
  for (int k0 = 0; k0 < K; k0 += 32) {
    const v16h b0 = frag_b_g(BT, K, n0, k0);
    const v16h b1 = frag_b_g(BT, K, n0 + 16, k0);
#pragma unroll
    for (int tm = 0; tm < 4; ++tm) {
      const v16h a = frag_a_g(A, K, m0 + tm * 16, k0);
      c[tm][0] = WMMA(a, b0, c[tm][0]);
      c[tm][1] = WMMA(a, b1, c[tm][1]);
    }
  }
  const int ln = l & 15;
  const int mh = (l & 16) ? 8 : 0;
#pragma unroll
  for (int tm = 0; tm < 4; ++tm)
#pragma unroll
    for (int tn = 0; tn < 2; ++tn) {
      const v8f acc = c[tm][tn];
      const int n = n0 + tn * 16 + ln;
      if (n >= N) continue;
      const float bv = bias ? bias[n] : 0.f;
#pragma unroll
      for (int r = 0; r < 8; ++r) {
        const int m = m0 + tm * 16 + mh + r;
        float v = acc[r] + bv;
        if (relu) v = fmaxf(v, 0.f);
        if (vt) {  // V-projection: store [B,H,HD,T] so AV GEMM is K-contiguous
          const int b = m >> 10, t = m & (TT - 1);
          const int h = n >> 6,  d = n & (HDD - 1);
          C16[(((size_t)(b * HH + h)) * HDD + d) * TT + t] = (_Float16)v;
        } else {
          const size_t idx = (size_t)m * N + n;
          if (C32) C32[idx] = v;
          if (C16) C16[idx] = (_Float16)v;
        }
      }
    }
}

// ------------------------------------------------------------------
// S[z,q,k] = dot(Q[b,q,h,:], K[b,k,h,:]) / 8
// K buffer is already [keypos, headdim] = K-contiguous B operand.
// grid = (T/32, T/256, B*H)
// ------------------------------------------------------------------
__launch_bounds__(128)
__global__ void attn_scores(const _Float16* __restrict__ Q,
                            const _Float16* __restrict__ Kc,
                            float* __restrict__ S) {
  const int w = threadIdx.x >> 5, l = threadIdx.x & 31;
  const int z = blockIdx.z, b = z >> 3, h = z & (HH - 1);
  const int q0 = blockIdx.y * 256 + w * 64;
  const int n0 = blockIdx.x * 32;
  const _Float16* Qb = Q  + (size_t)b * TT * DD + h * HDD;
  const _Float16* Kb = Kc + (size_t)b * TT * DD + h * HDD;
  v8f c[4][2] = {};
#pragma unroll
  for (int k0 = 0; k0 < HDD; k0 += 32) {
    const v16h b0 = frag_b_g(Kb, DD, n0, k0);
    const v16h b1 = frag_b_g(Kb, DD, n0 + 16, k0);
#pragma unroll
    for (int tm = 0; tm < 4; ++tm) {
      const v16h a = frag_a_g(Qb, DD, q0 + tm * 16, k0);
      c[tm][0] = WMMA(a, b0, c[tm][0]);
      c[tm][1] = WMMA(a, b1, c[tm][1]);
    }
  }
  float* Sb = S + (size_t)z * TT * TT;
  const int ln = l & 15;
  const int mh = (l & 16) ? 8 : 0;
#pragma unroll
  for (int tm = 0; tm < 4; ++tm)
#pragma unroll
    for (int tn = 0; tn < 2; ++tn) {
      const v8f acc = c[tm][tn];
      const int n = n0 + tn * 16 + ln;
#pragma unroll
      for (int r = 0; r < 8; ++r)
        Sb[(size_t)(q0 + tm * 16 + mh + r) * TT + n] = acc[r] * 0.125f;
    }
}

// ------------------------------------------------------------------
// O[b,q,h,:] = sum_k P[z,q,k] * V[b,k,h,:], with VT stored [z,HD,T]
// grid = (HD/32, T/256, B*H)
// ------------------------------------------------------------------
__launch_bounds__(128)
__global__ void attn_av(const _Float16* __restrict__ P,
                        const _Float16* __restrict__ VT,
                        _Float16* __restrict__ O) {
  const int w = threadIdx.x >> 5, l = threadIdx.x & 31;
  const int z = blockIdx.z, b = z >> 3, h = z & (HH - 1);
  const int m0 = blockIdx.y * 256 + w * 64;
  const int n0 = blockIdx.x * 32;
  const _Float16* Pb = P  + (size_t)z * TT * TT;
  const _Float16* Vb = VT + (size_t)z * HDD * TT;
  v8f c[4][2] = {};
  for (int k0 = 0; k0 < TT; k0 += 32) {
    const v16h b0 = frag_b_g(Vb, TT, n0, k0);
    const v16h b1 = frag_b_g(Vb, TT, n0 + 16, k0);
#pragma unroll
    for (int tm = 0; tm < 4; ++tm) {
      const v16h a = frag_a_g(Pb, TT, m0 + tm * 16, k0);
      c[tm][0] = WMMA(a, b0, c[tm][0]);
      c[tm][1] = WMMA(a, b1, c[tm][1]);
    }
  }
  const int ln = l & 15;
  const int mh = (l & 16) ? 8 : 0;
#pragma unroll
  for (int tm = 0; tm < 4; ++tm)
#pragma unroll
    for (int tn = 0; tn < 2; ++tn) {
      const v8f acc = c[tm][tn];
      const int n = n0 + tn * 16 + ln;
#pragma unroll
      for (int r = 0; r < 8; ++r) {
        const int m = m0 + tm * 16 + mh + r;
        O[(size_t)(b * TT + m) * DD + h * HDD + n] = (_Float16)acc[r];
      }
    }
}

// ------------------------------------------------------------------
// elementwise kernels
// ------------------------------------------------------------------
__launch_bounds__(256)
__global__ void embed_pe(const int* __restrict__ tok, const float* __restrict__ emb,
                         float* __restrict__ X32, _Float16* __restrict__ X16) {
  const int row = blockIdx.x;  // b*T + t
  const int t   = row % TT;
  const int id  = tok[row];
  for (int d = threadIdx.x; d < DD; d += 256) {
    float e   = (id == 0) ? 0.f : emb[(size_t)id * DD + d];  // padding_idx = 0
    float i2  = (float)(d >> 1);
    float ang = (float)t * powf(10000.f, -2.f * i2 / (float)DD);
    float pe  = (d & 1) ? cosf(ang) : sinf(ang);
    float v   = e + pe;
    X32[(size_t)row * DD + d] = v;
    X16[(size_t)row * DD + d] = (_Float16)v;
  }
}

__launch_bounds__(256)
__global__ void softmax_rows(const float* __restrict__ S,
                             _Float16* __restrict__ P, int causal) {
  __shared__ float red[256];
  const int row = blockIdx.x;  // z*T + q
  const int q   = row % TT;
  const float* s = S + (size_t)row * TT;
  _Float16*    p = P + (size_t)row * TT;
  const int tid   = threadIdx.x;
  const int limit = causal ? (q + 1) : TT;
  float mx = -3.402823466e38f;
  for (int k = tid; k < limit; k += 256) mx = fmaxf(mx, s[k]);
  red[tid] = mx; __syncthreads();
  for (int o = 128; o > 0; o >>= 1) {
    if (tid < o) red[tid] = fmaxf(red[tid], red[tid + o]);
    __syncthreads();
  }
  mx = red[0]; __syncthreads();
  float sum = 0.f;
  for (int k = tid; k < limit; k += 256) sum += expf(s[k] - mx);
  red[tid] = sum; __syncthreads();
  for (int o = 128; o > 0; o >>= 1) {
    if (tid < o) red[tid] += red[tid + o];
    __syncthreads();
  }
  const float inv = 1.f / red[0];
  for (int k = tid; k < TT; k += 256) {
    float v = (k < limit) ? expf(s[k] - mx) * inv : 0.f;
    p[k] = (_Float16)v;
  }
}

__launch_bounds__(256)
__global__ void add_ln(const float* __restrict__ X, const float* __restrict__ R,
                       const float* __restrict__ g, const float* __restrict__ bta,
                       float* __restrict__ Y32, _Float16* __restrict__ Y16) {
  __shared__ float red[256];
  const int row = blockIdx.x;
  const int tid = threadIdx.x;
  const float* x = X + (size_t)row * DD;
  const float* r = R + (size_t)row * DD;
  const float v0 = x[tid] + r[tid];
  const float v1 = x[tid + 256] + r[tid + 256];
  red[tid] = v0 + v1; __syncthreads();
  for (int o = 128; o > 0; o >>= 1) {
    if (tid < o) red[tid] += red[tid + o];
    __syncthreads();
  }
  const float mu = red[0] / (float)DD; __syncthreads();
  const float d0 = v0 - mu, d1 = v1 - mu;
  red[tid] = d0 * d0 + d1 * d1; __syncthreads();
  for (int o = 128; o > 0; o >>= 1) {
    if (tid < o) red[tid] += red[tid + o];
    __syncthreads();
  }
  const float rs = rsqrtf(red[0] / (float)DD + 1e-5f);
  const float y0 = d0 * rs * g[tid] + bta[tid];
  const float y1 = d1 * rs * g[tid + 256] + bta[tid + 256];
  Y32[(size_t)row * DD + tid]       = y0;
  Y32[(size_t)row * DD + tid + 256] = y1;
  Y16[(size_t)row * DD + tid]       = (_Float16)y0;
  Y16[(size_t)row * DD + tid + 256] = (_Float16)y1;
}

__launch_bounds__(256)
__global__ void argmax_rows(const float* __restrict__ Lg, float* __restrict__ out) {
  __shared__ float bv[256];
  __shared__ int   bi[256];
  const int row = blockIdx.x;
  const int tid = threadIdx.x;
  const float* x = Lg + (size_t)row * VOUT;
  float best = -3.402823466e38f; int bidx = 0;
  for (int k = tid; k < VOUT; k += 256) {
    const float v = x[k];
    if (v > best || (v == best && k < bidx)) { best = v; bidx = k; }
  }
  bv[tid] = best; bi[tid] = bidx; __syncthreads();
  for (int o = 128; o > 0; o >>= 1) {
    if (tid < o) {
      if (bv[tid + o] > bv[tid] ||
          (bv[tid + o] == bv[tid] && bi[tid + o] < bi[tid])) {
        bv[tid] = bv[tid + o]; bi[tid] = bi[tid + o];
      }
    }
    __syncthreads();
  }
  if (tid == 0) out[row] = (float)(bi[0] + 2);
}

// ------------------------------------------------------------------
// host orchestration
// ------------------------------------------------------------------
extern "C" void kernel_launch(void* const* d_in, const int* in_sizes, int n_in,
                              void* d_out, int out_size, void* d_ws, size_t ws_size,
                              hipStream_t stream) {
  const int*   enc_in   = (const int*)  d_in[0];
  const int*   dec_in   = (const int*)  d_in[1];
  const float* emb      = (const float*)d_in[2];
  const float* enc_wq   = (const float*)d_in[3];
  const float* enc_wk   = (const float*)d_in[4];
  const float* enc_wv   = (const float*)d_in[5];
  const float* enc_wo   = (const float*)d_in[6];
  const float* enc_ln1g = (const float*)d_in[7];
  const float* enc_ln1b = (const float*)d_in[8];
  const float* enc_w1   = (const float*)d_in[9];
  const float* enc_b1   = (const float*)d_in[10];
  const float* enc_w2   = (const float*)d_in[11];
  const float* enc_b2   = (const float*)d_in[12];
  const float* enc_ln2g = (const float*)d_in[13];
  const float* enc_ln2b = (const float*)d_in[14];
  const float* dsa_wq   = (const float*)d_in[15];
  const float* dsa_wk   = (const float*)d_in[16];
  const float* dsa_wv   = (const float*)d_in[17];
  const float* dsa_wo   = (const float*)d_in[18];
  const float* dec_wq   = (const float*)d_in[19];
  const float* dec_wk   = (const float*)d_in[20];
  const float* dec_wv   = (const float*)d_in[21];
  const float* dec_wo   = (const float*)d_in[22];
  const float* dec_ln1g = (const float*)d_in[23];
  const float* dec_ln1b = (const float*)d_in[24];
  const float* dec_w1   = (const float*)d_in[25];
  const float* dec_b1   = (const float*)d_in[26];
  const float* dec_w2   = (const float*)d_in[27];
  const float* dec_b2   = (const float*)d_in[28];
  const float* dec_ln2g = (const float*)d_in[29];
  const float* dec_ln2b = (const float*)d_in[30];
  const float* out_w    = (const float*)d_in[31];
  const float* out_b    = (const float*)d_in[32];
  (void)in_sizes; (void)n_in; (void)out_size; (void)ws_size;

  char* ws = (char*)d_ws;
  size_t off = 0;
  auto alloc = [&](size_t bytes) -> char* {
    char* p = ws + off;
    off += (bytes + 255) & ~(size_t)255;
    return p;
  };
  // activations
  float*    xe32   = (float*)    alloc((size_t)MM * DD * 4);
  _Float16* xe16   = (_Float16*) alloc((size_t)MM * DD * 2);
  float*    xd32   = (float*)    alloc((size_t)MM * DD * 4);
  _Float16* xd16   = (_Float16*) alloc((size_t)MM * DD * 2);
  _Float16* q16    = (_Float16*) alloc((size_t)MM * DD * 2);
  _Float16* k16    = (_Float16*) alloc((size_t)MM * DD * 2);
  _Float16* vt16   = (_Float16*) alloc((size_t)MM * DD * 2);   // [B,H,HD,T]
  _Float16* ao16   = (_Float16*) alloc((size_t)MM * DD * 2);
  float*    mha32  = (float*)    alloc((size_t)MM * DD * 4);
  float*    so32   = (float*)    alloc((size_t)MM * DD * 4);
  _Float16* so16   = (_Float16*) alloc((size_t)MM * DD * 2);
  float*    h32    = (float*)    alloc((size_t)MM * DD * 4);
  _Float16* h16    = (_Float16*) alloc((size_t)MM * DD * 2);
  _Float16* ff1_16 = (_Float16*) alloc((size_t)MM * FFD * 2);
  float*    ff2_32 = (float*)    alloc((size_t)MM * DD * 4);
  float*    scores = (float*)    alloc((size_t)BB * HH * TT * TT * 4);
  _Float16* probs  = (_Float16*) alloc((size_t)BB * HH * TT * TT * 2);
  float*    logits = (float*)    alloc((size_t)MM * VOUT * 4);
  // transposed f16 weights ([N,K] layouts; whole set ~92MB -> lives in L2)
  const size_t szDD = (size_t)LL * DD * DD;
  const size_t szDF = (size_t)LL * DD * FFD;
  _Float16* encqT = (_Float16*)alloc(szDD * 2);
  _Float16* enckT = (_Float16*)alloc(szDD * 2);
  _Float16* encvT = (_Float16*)alloc(szDD * 2);
  _Float16* encoT = (_Float16*)alloc(szDD * 2);
  _Float16* dsaqT = (_Float16*)alloc(szDD * 2);
  _Float16* dsakT = (_Float16*)alloc(szDD * 2);
  _Float16* dsavT = (_Float16*)alloc(szDD * 2);
  _Float16* dsaoT = (_Float16*)alloc(szDD * 2);
  _Float16* decqT = (_Float16*)alloc(szDD * 2);
  _Float16* deckT = (_Float16*)alloc(szDD * 2);
  _Float16* decvT = (_Float16*)alloc(szDD * 2);
  _Float16* decoT = (_Float16*)alloc(szDD * 2);
  _Float16* encw1T = (_Float16*)alloc(szDF * 2);  // [FF,D] per layer
  _Float16* encw2T = (_Float16*)alloc(szDF * 2);  // [D,FF] per layer
  _Float16* decw1T = (_Float16*)alloc(szDF * 2);
  _Float16* decw2T = (_Float16*)alloc(szDF * 2);
  _Float16* outwT  = (_Float16*)alloc((size_t)VOUTP * DD * 2);  // [VOUTP,D]

  const dim3 blk128(128), blk256(256);
  auto tpose = [&](const float* W, _Float16* WT, int K, int N, int Npad) {
    transpose_f16<<<dim3(Npad / 32, K / 32), blk256, 0, stream>>>(W, WT, K, N);
  };
  // one-time (per launch) weight transposition
  for (int l = 0; l < LL; ++l) {
    const size_t oD = (size_t)l * DD * DD, oF = (size_t)l * DD * FFD;
    tpose(enc_wq + oD, encqT + oD, DD, DD, DD);
    tpose(enc_wk + oD, enckT + oD, DD, DD, DD);
    tpose(enc_wv + oD, encvT + oD, DD, DD, DD);
    tpose(enc_wo + oD, encoT + oD, DD, DD, DD);
    tpose(dsa_wq + oD, dsaqT + oD, DD, DD, DD);
    tpose(dsa_wk + oD, dsakT + oD, DD, DD, DD);
    tpose(dsa_wv + oD, dsavT + oD, DD, DD, DD);
    tpose(dsa_wo + oD, dsaoT + oD, DD, DD, DD);
    tpose(dec_wq + oD, decqT + oD, DD, DD, DD);
    tpose(dec_wk + oD, deckT + oD, DD, DD, DD);
    tpose(dec_wv + oD, decvT + oD, DD, DD, DD);
    tpose(dec_wo + oD, decoT + oD, DD, DD, DD);
    tpose(enc_w1 + oF, encw1T + oF, DD, FFD, FFD);   // [D,FF] -> [FF,D]
    tpose(enc_w2 + oF, encw2T + oF, FFD, DD, DD);    // [FF,D] -> [D,FF]
    tpose(dec_w1 + oF, decw1T + oF, DD, FFD, FFD);
    tpose(dec_w2 + oF, decw2T + oF, FFD, DD, DD);
  }
  tpose(out_w, outwT, DD, VOUT, VOUTP);

  auto gemm = [&](const _Float16* A, const _Float16* BT, const float* bias,
                  float* C32, _Float16* C16, int N, int Npad, int K,
                  int relu, int vt) {
    gemm_wmma<<<dim3(Npad / 32, MM / 256), blk128, 0, stream>>>(
        A, BT, bias, C32, C16, N, K, relu, vt);
  };
  auto mha = [&](const _Float16* xq, const _Float16* xk, const _Float16* xv,
                 const _Float16* wqT, const _Float16* wkT, const _Float16* wvT,
                 const _Float16* woT, int causal, float* out32, _Float16* out16) {
    gemm(xq, wqT, nullptr, nullptr, q16, DD, DD, DD, 0, 0);
    gemm(xk, wkT, nullptr, nullptr, k16, DD, DD, DD, 0, 0);
    gemm(xv, wvT, nullptr, nullptr, vt16, DD, DD, DD, 0, 1);  // head-transposed
    attn_scores<<<dim3(TT / 32, TT / 256, BB * HH), blk128, 0, stream>>>(
        q16, k16, scores);
    softmax_rows<<<BB * HH * TT, blk256, 0, stream>>>(scores, probs, causal);
    attn_av<<<dim3(HDD / 32, TT / 256, BB * HH), blk128, 0, stream>>>(
        probs, vt16, ao16);
    gemm(ao16, woT, nullptr, out32, out16, DD, DD, DD, 0, 0);
  };

  // embeddings + positional encoding
  embed_pe<<<MM, blk256, 0, stream>>>(enc_in, emb, xe32, xe16);
  embed_pe<<<MM, blk256, 0, stream>>>(dec_in, emb, xd32, xd16);

  // ---- encoder ----
  for (int l = 0; l < LL; ++l) {
    const size_t oD = (size_t)l * DD * DD, oF = (size_t)l * DD * FFD;
    mha(xe16, xe16, xe16, encqT + oD, enckT + oD, encvT + oD, encoT + oD,
        0, mha32, nullptr);
    add_ln<<<MM, blk256, 0, stream>>>(mha32, xe32,
        enc_ln1g + (size_t)l * DD, enc_ln1b + (size_t)l * DD, h32, h16);
    gemm(h16, encw1T + oF, enc_b1 + (size_t)l * FFD, nullptr, ff1_16,
         FFD, FFD, DD, 1, 0);
    gemm(ff1_16, encw2T + oF, enc_b2 + (size_t)l * DD, ff2_32, nullptr,
         DD, DD, FFD, 0, 0);
    add_ln<<<MM, blk256, 0, stream>>>(ff2_32, h32,
        enc_ln2g + (size_t)l * DD, enc_ln2b + (size_t)l * DD, xe32, xe16);
  }

  // ---- decoder ----
  for (int l = 0; l < LL; ++l) {
    const size_t oD = (size_t)l * DD * DD, oF = (size_t)l * DD * FFD;
    // masked self-attention (no residual, no LN)
    mha(xd16, xd16, xd16, dsaqT + oD, dsakT + oD, dsavT + oD, dsaoT + oD,
        1, so32, so16);
    // cross block: Q=K=enc_x, V=self_out; residual on V input
    mha(xe16, xe16, so16, decqT + oD, deckT + oD, decvT + oD, decoT + oD,
        0, mha32, nullptr);
    add_ln<<<MM, blk256, 0, stream>>>(mha32, so32,
        dec_ln1g + (size_t)l * DD, dec_ln1b + (size_t)l * DD, h32, h16);
    gemm(h16, decw1T + oF, dec_b1 + (size_t)l * FFD, nullptr, ff1_16,
         FFD, FFD, DD, 1, 0);
    gemm(ff1_16, decw2T + oF, dec_b2 + (size_t)l * DD, ff2_32, nullptr,
         DD, DD, FFD, 0, 0);
    add_ln<<<MM, blk256, 0, stream>>>(ff2_32, h32,
        dec_ln2g + (size_t)l * DD, dec_ln2b + (size_t)l * DD, xd32, xd16);
  }

  // logits + argmax (+2); softmax is monotone so argmax(logits) == argmax(probs)
  gemm(xd16, outwT, out_b, logits, nullptr, VOUT, VOUTP, DD, 0, 0);
  argmax_rows<<<MM, blk256, 0, stream>>>(logits, (float*)d_out);
}
